// Attn_42425686950260
// MI455X (gfx1250) — matrix operand order repaired and checked by reading, not compiled, not run on hardware
//
#include <hip/hip_runtime.h>
#include <hip/hip_bf16.h>

// ---------------------------------------------------------------------------
// HEPT block-local attention layer for MI455X (gfx1250, wave32, WMMA).
// All matmuls use v_wmma_f32_16x16x32_f16 (f16 in, f32 accumulate).
// ---------------------------------------------------------------------------

#define NTOK   65536
#define HEADS  8
#define DIM    32
#define HD     256      // HEADS*DIM
#define BLK    256      // attention block
#define NB     256      // NTOK/BLK

typedef __attribute__((ext_vector_type(16))) _Float16 v16h;
typedef __attribute__((ext_vector_type(8)))  _Float16 v8h;
typedef __attribute__((ext_vector_type(8)))  float    v8f;

static __device__ __forceinline__ v8h ld8h(const _Float16* p) {
    return *(const v8h*)p;
}
static __device__ __forceinline__ v16h mk16(v8h lo, v8h hi) {
    v16h r;
#pragma unroll
    for (int i = 0; i < 8; ++i) { r[i] = lo[i]; r[i + 8] = hi[i]; }
    return r;
}
static __device__ __forceinline__ v8f wmma_f16(v16h a, v16h b, v8f c) {
    // (neg_a, A, neg_b, B, c_mod, C, reuse_a, reuse_b)
    return __builtin_amdgcn_wmma_f32_16x16x32_f16(false, a, false, b,
                                                  (short)0, c, false, false);
}

// A-fragment (16x32 f16, M x K): lane holds row = lane&15; K elements
// {kh*16 + j} for j<16, kh = lane>>4.
static __device__ __forceinline__ v16h load_afrag(const _Float16* row, int kh) {
    return mk16(ld8h(row + kh * 16), ld8h(row + kh * 16 + 8));
}
// B-fragment (32x16 f16, K x N) from a row-major matrix indexed "row = n"
// (B = M^T): lane holds col n = lane&15, elements j: K = kh*16 + j.
static __device__ __forceinline__ v16h load_bfrag_row(const _Float16* row, int kh) {
    return mk16(ld8h(row + kh * 16), ld8h(row + kh * 16 + 8));
}
// Packed B-fragment (1 KB per fragment, lane-major 32B per lane).
static __device__ __forceinline__ v16h load_bfrag_packed(const _Float16* frag, int lane) {
    const _Float16* p = frag + lane * 16;
    return mk16(ld8h(p), ld8h(p + 8));
}

// ---------------------------------------------------------------------------
// Workspace layout (bytes).
// ---------------------------------------------------------------------------
#define OFF_WFRAG  ((size_t)0)                          // 68 * 1024 B packed B-frags
#define OFF_W2     ((size_t)131072)                     // 16 floats (rpe weights)
#define OFF_KEYS   (OFF_W2 + 1024)                      // N f32 sort keys
#define OFF_ORDER  (OFF_KEYS + (size_t)NTOK * 4)        // N i32 permutation
#define OFF_PS     (OFF_ORDER + (size_t)NTOK * 4)       // N*2 f32 sorted coords
#define OFF_QS     (OFF_PS + (size_t)NTOK * 8)          // N*256 f16 Q (sorted)
#define OFF_KS     (OFF_QS + (size_t)NTOK * HD * 2)     // N*256 f16 K (sorted)
#define OFF_VT     (OFF_KS + (size_t)NTOK * HD * 2)     // 256*N f16 V^T (sorted)
#define OFF_OUTS   (OFF_VT + (size_t)NTOK * HD * 2)     // N*256 f16 attn out (sorted)
#define OFF_X2     (OFF_OUTS + (size_t)NTOK * HD * 2)   // N*32 f32 post-attn residual

// ---------------------------------------------------------------------------
// 1) Pack weight matrices into WMMA B-fragments.
//    frag 0..47   : wq/wk/wv  (proj = f>>4, ntile = f&15), B[k][n] = W[n*32+k]
//    frag 48..63  : w_out     (kt = g>>1, nt = g&1),       B[k][n] = W[n*256+k]
//    frag 64..65  : ff_w1,  frag 66..67 : ff_w2
// ---------------------------------------------------------------------------
__global__ void pack_weights_kernel(const float* __restrict__ wq,
                                    const float* __restrict__ wk,
                                    const float* __restrict__ wv,
                                    const float* __restrict__ w_out,
                                    const float* __restrict__ ffw1,
                                    const float* __restrict__ ffw2,
                                    _Float16* __restrict__ wfrag) {
    int f = blockIdx.x;
    int lane = threadIdx.x;
    int lm = lane & 15, kh = lane >> 4;
    _Float16* dst = wfrag + (size_t)f * 512 + lane * 16;
#pragma unroll
    for (int j = 0; j < 16; ++j) {
        float val;
        if (f < 48) {
            int proj = f >> 4, nt = f & 15;
            int n = nt * 16 + lm, k = kh * 16 + j;
            const float* W = (proj == 0) ? wq : ((proj == 1) ? wk : wv);
            val = W[n * 32 + k];
        } else if (f < 64) {
            int g = f - 48, kt = g >> 1, nt = g & 1;
            int n = nt * 16 + lm, k = kt * 32 + kh * 16 + j;
            val = w_out[n * 256 + k];
        } else if (f < 66) {
            int nt = f - 64;
            val = ffw1[(nt * 16 + lm) * 32 + kh * 16 + j];
        } else {
            int nt = f - 66;
            val = ffw2[(nt * 16 + lm) * 32 + kh * 16 + j];
        }
        dst[j] = (_Float16)val;
    }
}

// ---------------------------------------------------------------------------
// 2) RPE distance weights: w2[h,c] = mean_{d,w} W[h,d,c,w]^2   (8x2 floats)
// ---------------------------------------------------------------------------
__global__ void compute_w2_kernel(const float* __restrict__ w_rpe,
                                  float* __restrict__ w2) {
    int h = blockIdx.x;         // 8 blocks
    int d = threadIdx.x;        // 32 lanes = one wave = one head's D
    const float* row = w_rpe + (size_t)(h * 32 + d) * 16;
    float s0 = 0.f, s1 = 0.f;
#pragma unroll
    for (int wi = 0; wi < 8; ++wi) {
        s0 += row[wi] * row[wi];
        s1 += row[8 + wi] * row[8 + wi];
    }
#pragma unroll
    for (int m = 1; m < 32; m <<= 1) {
        s0 += __shfl_xor(s0, m, 32);
        s1 += __shfl_xor(s1, m, 32);
    }
    if (d == 0) {
        w2[h * 2 + 0] = s0 * (1.0f / 256.0f);
        w2[h * 2 + 1] = s1 * (1.0f / 256.0f);
    }
}

// ---------------------------------------------------------------------------
// 3) Bitonic argsort of coords[:,0]  (keys rebuilt each launch -> deterministic)
//    Global passes for j > 256; all passes j <= 256 fused in one LDS kernel
//    (aligned 512-element windows are closed under i ^ j for j < 512).
// ---------------------------------------------------------------------------
__global__ void sort_init_kernel(const float* __restrict__ coords,
                                 float* __restrict__ keys,
                                 int* __restrict__ order) {
    int i = blockIdx.x * blockDim.x + threadIdx.x;
    if (i < NTOK) { keys[i] = coords[3 * i]; order[i] = i; }
}

__global__ void bitonic_pass_kernel(float* __restrict__ keys,
                                    int* __restrict__ order, int j, int k) {
    int i = blockIdx.x * blockDim.x + threadIdx.x;
    int p = i ^ j;
    if (p > i) {
        bool up = ((i & k) == 0);
        float ki = keys[i], kp = keys[p];
        if ((ki > kp) == up) {
            keys[i] = kp; keys[p] = ki;
            int t = order[i]; order[i] = order[p]; order[p] = t;
        }
    }
}

__global__ void bitonic_fused_kernel(float* __restrict__ keys,
                                     int* __restrict__ order,
                                     int k, int jstart) {
    __shared__ float sk[512];
    __shared__ int   si[512];
    int base = blockIdx.x * 512;
    int t = threadIdx.x;
    sk[t] = keys[base + t];         sk[t + 256] = keys[base + t + 256];
    si[t] = order[base + t];        si[t + 256] = order[base + t + 256];
    __syncthreads();
    for (int j = jstart; j > 0; j >>= 1) {
        int i = ((t & ~(j - 1)) << 1) | (t & (j - 1));   // low index of pair
        int p = i | j;
        bool up = (((base + i) & k) == 0);
        float ki = sk[i], kp = sk[p];
        if ((ki > kp) == up) {
            sk[i] = kp; sk[p] = ki;
            int tmp = si[i]; si[i] = si[p]; si[p] = tmp;
        }
        __syncthreads();
    }
    keys[base + t] = sk[t];         keys[base + t + 256] = sk[t + 256];
    order[base + t] = si[t];        order[base + t + 256] = si[t + 256];
}

__global__ void gather_coords_kernel(const int* __restrict__ order,
                                     const float* __restrict__ coords,
                                     float* __restrict__ p_s) {
    int s = blockIdx.x * blockDim.x + threadIdx.x;
    if (s < NTOK) {
        int t = order[s];
        p_s[2 * s + 0] = coords[3 * t + 1];
        p_s[2 * s + 1] = coords[3 * t + 2];
    }
}

// ---------------------------------------------------------------------------
// 4) Fused LayerNorm + QKV projection, gathered into sorted order.
//    256 threads / 64 tokens. Columns (proj,ntile) are assigned as
//    col = cc*8 + wave with cc unrolled 0..5 so proj = cc>>1 is a
//    compile-time constant (no divergent store epilogue).
//    Q scaled by 1/sqrt(D); V stored transposed (vt[h*32+d][s]) so the
//    8 accumulator rows per lane form one contiguous 16B store.
// ---------------------------------------------------------------------------
__global__ void ln_qkv_kernel(const float* __restrict__ x,
                              const int* __restrict__ order,
                              const float* __restrict__ g1,
                              const float* __restrict__ be1,
                              const _Float16* __restrict__ wfrag,
                              _Float16* __restrict__ q_s,
                              _Float16* __restrict__ k_s,
                              _Float16* __restrict__ vt) {
    __shared__ _Float16 xn[64 * 32];
    int wg = blockIdx.x, tid = threadIdx.x;

    // --- LayerNorm: 4 threads per row, 8 features each, shuffle-combine ---
    {
        int r = tid >> 2, q4 = tid & 3;
        const float* xr = x + (size_t)order[wg * 64 + r] * 32;
        float buf[8], sum = 0.f, sq = 0.f;
#pragma unroll
        for (int e = 0; e < 8; ++e) {
            float v = xr[q4 * 8 + e];
            buf[e] = v; sum += v; sq += v * v;
        }
        sum += __shfl_xor(sum, 1, 4); sum += __shfl_xor(sum, 2, 4);
        sq  += __shfl_xor(sq, 1, 4);  sq  += __shfl_xor(sq, 2, 4);
        float mu = sum * (1.0f / 32.0f);
        float var = sq * (1.0f / 32.0f) - mu * mu;
        float rstd = rsqrtf(var + 1e-5f);
#pragma unroll
        for (int e = 0; e < 8; ++e) {
            int dd = q4 * 8 + e;
            xn[r * 32 + dd] = (_Float16)((buf[e] - mu) * rstd * g1[dd] + be1[dd]);
        }
    }
    __syncthreads();

    int w = tid >> 5, lane = tid & 31, lm = lane & 15, kh = lane >> 4;
    for (int mt = 0; mt < 4; ++mt) {
        v16h a = load_afrag(xn + (mt * 16 + lm) * 32, kh);   // reused 6x
        int srow0 = wg * 64 + mt * 16 + 8 * kh;
#pragma unroll
        for (int cc = 0; cc < 6; ++cc) {
            const int proj = cc >> 1;                 // compile-time constant
            int nt = (cc & 1) * 8 + w;
            v16h b = load_bfrag_packed(wfrag + (size_t)(proj * 16 + nt) * 512, lane);
            v8f z = {};
            v8f acc = wmma_f16(a, b, z);
            int col = nt * 16 + lm;
            if (proj == 0) {
                _Float16* dst = q_s + (size_t)srow0 * HD + col;
#pragma unroll
                for (int r = 0; r < 8; ++r)
                    dst[r * HD] = (_Float16)(acc[r] * 0.17677669529663687f);
            } else if (proj == 1) {
                _Float16* dst = k_s + (size_t)srow0 * HD + col;
#pragma unroll
                for (int r = 0; r < 8; ++r)
                    dst[r * HD] = (_Float16)acc[r];
            } else {
                v8h hv;
#pragma unroll
                for (int r = 0; r < 8; ++r) hv[r] = (_Float16)acc[r];
                *(v8h*)(vt + (size_t)col * NTOK + srow0) = hv;
            }
        }
    }
}

// ---------------------------------------------------------------------------
// 5) Block-local attention. grid = (NB, HEADS), 128 threads (4 waves).
//    Each wave handles 4 query tiles of 16 rows: 16 QK^T WMMAs, register
//    softmax with RPE bias (16-lane shuffle reductions), P staged f16 in a
//    per-wave LDS strip (D->A layout crossing), 16 PV WMMAs, 1/rowsum scale.
// ---------------------------------------------------------------------------
__global__ void attention_kernel(const _Float16* __restrict__ q_s,
                                 const _Float16* __restrict__ k_s,
                                 const _Float16* __restrict__ vt,
                                 const float* __restrict__ p_s,
                                 const float* __restrict__ w2,
                                 _Float16* __restrict__ out_s) {
    __shared__ _Float16 strip[4][16 * 256];   // 32 KB: per-wave P tile

    int blk = blockIdx.x, h = blockIdx.y;
    int tid = threadIdx.x;
    int w = tid >> 5, lane = tid & 31, lm = lane & 15, kh = lane >> 4;
    int s0 = blk * BLK;
    float w20 = w2[h * 2], w21 = w2[h * 2 + 1];

    for (int mseg = 0; mseg < 4; ++mseg) {
        int mt = mseg * 4 + w;

        // A fragment of Q for this 16-row tile (L2-resident f16 rows).
        const _Float16* qrow = q_s + (size_t)(s0 + mt * 16 + lm) * HD + h * 32;
        v16h aQ = load_afrag(qrow, kh);

        // Query-side coords for this lane's 8 accumulator rows.
        const float* pir = p_s + 2 * (s0 + mt * 16 + 8 * kh);
        float pi0[8], pi1[8];
#pragma unroll
        for (int r = 0; r < 8; ++r) { pi0[r] = pir[2 * r]; pi1[r] = pir[2 * r + 1]; }

        // ---- S = Q K^T (16 x 256) ----
        v8f acc[16];
#pragma unroll
        for (int nt = 0; nt < 16; ++nt) {
            const _Float16* krow = k_s + (size_t)(s0 + nt * 16 + lm) * HD + h * 32;
            v16h bK = load_bfrag_row(krow, kh);
            v8f z = {};
            acc[nt] = wmma_f16(aQ, bK, z);
        }

        // ---- RPE bias + row max ----
        float mrow[8];
#pragma unroll
        for (int r = 0; r < 8; ++r) mrow[r] = -1e30f;
#pragma unroll
        for (int nt = 0; nt < 16; ++nt) {
            int j = s0 + nt * 16 + lm;
            float pj0 = p_s[2 * j], pj1 = p_s[2 * j + 1];
#pragma unroll
            for (int r = 0; r < 8; ++r) {
                float d0 = pi0[r] - pj0, d1 = pi1[r] - pj1;
                float sc = acc[nt][r] - (w20 * d0 * d0 + w21 * d1 * d1);
                acc[nt][r] = sc;
                mrow[r] = fmaxf(mrow[r], sc);
            }
        }
#pragma unroll
        for (int m = 1; m < 16; m <<= 1)
#pragma unroll
            for (int r = 0; r < 8; ++r)
                mrow[r] = fmaxf(mrow[r], __shfl_xor(mrow[r], m, 32));

        // ---- exp + row sum; stage P (f16) to LDS in memory row-major ----
        float rs[8];
#pragma unroll
        for (int r = 0; r < 8; ++r) rs[r] = 0.f;
#pragma unroll
        for (int nt = 0; nt < 16; ++nt) {
            _Float16* sp = &strip[w][(8 * kh) * 256 + nt * 16 + lm];
#pragma unroll
            for (int r = 0; r < 8; ++r) {
                float p = __expf(acc[nt][r] - mrow[r]);
                rs[r] += p;
                sp[r * 256] = (_Float16)p;
            }
        }
#pragma unroll
        for (int m = 1; m < 16; m <<= 1)
#pragma unroll
            for (int r = 0; r < 8; ++r)
                rs[r] += __shfl_xor(rs[r], m, 32);

        __syncthreads();   // publish P strips (uniform across workgroup)

        // ---- O = P V  (16 x 32): K = 256 -> 8 k-steps x 2 n-tiles ----
        v8f o0 = {}, o1 = {};
        const _Float16* prow = strip[w] + lm * 256;
#pragma unroll
        for (int kt = 0; kt < 8; ++kt) {
            v16h aP = load_afrag(prow + kt * 32, kh);
            const _Float16* v0 = vt + (size_t)(h * 32 + lm) * NTOK + s0 + kt * 32;
            const _Float16* v1 = vt + (size_t)(h * 32 + 16 + lm) * NTOK + s0 + kt * 32;
            o0 = wmma_f16(aP, load_bfrag_row(v0, kh), o0);
            o1 = wmma_f16(aP, load_bfrag_row(v1, kh), o1);
        }

        // ---- scale by 1/rowsum, store f16 (immediate-offset stores) ----
        _Float16* dst = out_s + (size_t)(s0 + mt * 16 + 8 * kh) * HD + h * 32 + lm;
#pragma unroll
        for (int r = 0; r < 8; ++r) {
            float is = 1.0f / rs[r];
            dst[r * HD]      = (_Float16)(o0[r] * is);
            dst[r * HD + 16] = (_Float16)(o1[r] * is);
        }
        __syncthreads();   // strips reusable next mseg
    }
}

// ---------------------------------------------------------------------------
// 6) Output projection (256 -> 32) + scatter-by-order residual:
//    x2[t] = x[t] + out_s[s] @ w_out^T + b_out,  t = order[s].
// ---------------------------------------------------------------------------
__global__ void out_proj_kernel(const _Float16* __restrict__ out_s,
                                const _Float16* __restrict__ wfrag,
                                const float* __restrict__ b_out,
                                const float* __restrict__ x,
                                const int* __restrict__ order,
                                float* __restrict__ x2) {
    int wg = blockIdx.x, tid = threadIdx.x;
    int w = tid >> 5, lane = tid & 31, lm = lane & 15, kh = lane >> 4;
    int sbase = wg * 64 + w * 16;

    v8f a0 = {}, a1 = {};
    const _Float16* arow = out_s + (size_t)(sbase + lm) * HD;
#pragma unroll
    for (int kt = 0; kt < 8; ++kt) {
        v16h aO = load_afrag(arow + kt * 32, kh);
        a0 = wmma_f16(aO, load_bfrag_packed(wfrag + (size_t)(48 + kt * 2) * 512, lane), a0);
        a1 = wmma_f16(aO, load_bfrag_packed(wfrag + (size_t)(49 + kt * 2) * 512, lane), a1);
    }
    float bo0 = b_out[lm], bo1 = b_out[16 + lm];
    const int* ord = order + sbase + 8 * kh;
#pragma unroll
    for (int r = 0; r < 8; ++r) {
        int t = ord[r];
        const float* xr = x + (size_t)t * 32;
        float* x2r = x2 + (size_t)t * 32;
        x2r[lm]      = xr[lm]      + a0[r] + bo0;
        x2r[16 + lm] = xr[16 + lm] + a1[r] + bo1;
    }
}

// ---------------------------------------------------------------------------
// 7) FFN: out = x2 + relu(LN(x2) @ ff_w1^T + b1) @ ff_w2^T + b2
//    128 threads / 64 tokens; ReLU output crosses D->A layout via LDS strip.
// ---------------------------------------------------------------------------
__global__ void ffn_kernel(const float* __restrict__ x2,
                           const float* __restrict__ g2,
                           const float* __restrict__ be2,
                           const _Float16* __restrict__ wfrag,
                           const float* __restrict__ ff_b1,
                           const float* __restrict__ ff_b2,
                           float* __restrict__ out) {
    __shared__ _Float16 xn[64 * 32];
    __shared__ _Float16 h1s[4][16 * 32];
    int wg = blockIdx.x, tid = threadIdx.x;

    // --- LayerNorm: 2 threads per row, 16 features each ---
    {
        int r = tid >> 1, hf = tid & 1;
        const float* xr = x2 + (size_t)(wg * 64 + r) * 32;
        float buf[16], sum = 0.f, sq = 0.f;
#pragma unroll
        for (int e = 0; e < 16; ++e) {
            float v = xr[hf * 16 + e];
            buf[e] = v; sum += v; sq += v * v;
        }
        sum += __shfl_xor(sum, 1, 2);
        sq  += __shfl_xor(sq, 1, 2);
        float mu = sum * (1.0f / 32.0f);
        float var = sq * (1.0f / 32.0f) - mu * mu;
        float rstd = rsqrtf(var + 1e-5f);
#pragma unroll
        for (int e = 0; e < 16; ++e) {
            int dd = hf * 16 + e;
            xn[r * 32 + dd] = (_Float16)((buf[e] - mu) * rstd * g2[dd] + be2[dd]);
        }
    }
    __syncthreads();

    int w = tid >> 5, lane = tid & 31, lm = lane & 15, kh = lane >> 4;
    int mt = w;

    // --- h1 = relu(xn @ ff_w1^T + b1) -> LDS strip (f16) ---
    v16h aX = load_afrag(xn + (mt * 16 + lm) * 32, kh);
#pragma unroll
    for (int nt = 0; nt < 2; ++nt) {
        v8f z = {};
        v8f acc = wmma_f16(aX, load_bfrag_packed(wfrag + (size_t)(64 + nt) * 512, lane), z);
        int n = nt * 16 + lm;
        float b1 = ff_b1[n];
        _Float16* hp = &h1s[w][(8 * kh) * 32 + n];
#pragma unroll
        for (int r = 0; r < 8; ++r)
            hp[r * 32] = (_Float16)fmaxf(acc[r] + b1, 0.0f);
    }
    __syncthreads();

    // --- out = x2 + h1 @ ff_w2^T + b2 ---
    v16h aH = load_afrag(h1s[w] + lm * 32, kh);
#pragma unroll
    for (int nt = 0; nt < 2; ++nt) {
        v8f z = {};
        v8f acc = wmma_f16(aH, load_bfrag_packed(wfrag + (size_t)(66 + nt) * 512, lane), z);
        int n = nt * 16 + lm;
        float b2 = ff_b2[n];
        const float* x2r = x2 + (size_t)(wg * 64 + mt * 16 + 8 * kh) * 32 + n;
        float* outr = out + (size_t)(wg * 64 + mt * 16 + 8 * kh) * 32 + n;
#pragma unroll
        for (int r = 0; r < 8; ++r)
            outr[r * 32] = x2r[r * 32] + acc[r] + b2;
    }
}

// ---------------------------------------------------------------------------
// Host launcher
// ---------------------------------------------------------------------------
extern "C" void kernel_launch(void* const* d_in, const int* in_sizes, int n_in,
                              void* d_out, int out_size, void* d_ws, size_t ws_size,
                              hipStream_t stream) {
    (void)in_sizes; (void)n_in; (void)out_size; (void)ws_size;

    const float* x      = (const float*)d_in[0];
    const float* coords = (const float*)d_in[1];
    const float* wq     = (const float*)d_in[2];
    const float* wk     = (const float*)d_in[3];
    const float* wv     = (const float*)d_in[4];
    const float* w_rpe  = (const float*)d_in[5];
    const float* w_out  = (const float*)d_in[6];
    const float* b_out  = (const float*)d_in[7];
    const float* g1     = (const float*)d_in[8];
    const float* be1    = (const float*)d_in[9];
    const float* g2     = (const float*)d_in[10];
    const float* be2    = (const float*)d_in[11];
    const float* ff_w1  = (const float*)d_in[12];
    const float* ff_b1  = (const float*)d_in[13];
    const float* ff_w2  = (const float*)d_in[14];
    const float* ff_b2  = (const float*)d_in[15];
    float* y            = (float*)d_out;

    char* ws = (char*)d_ws;
    _Float16* wfrag = (_Float16*)(ws + OFF_WFRAG);
    float*    w2    = (float*)(ws + OFF_W2);
    float*    keys  = (float*)(ws + OFF_KEYS);
    int*      order = (int*)(ws + OFF_ORDER);
    float*    p_s   = (float*)(ws + OFF_PS);
    _Float16* q_s   = (_Float16*)(ws + OFF_QS);
    _Float16* k_s   = (_Float16*)(ws + OFF_KS);
    _Float16* vt    = (_Float16*)(ws + OFF_VT);
    _Float16* out_s = (_Float16*)(ws + OFF_OUTS);
    float*    x2    = (float*)(ws + OFF_X2);

    // 1) pack weights + 2) rpe weights (independent of sort)
    pack_weights_kernel<<<68, 32, 0, stream>>>(wq, wk, wv, w_out, ff_w1, ff_w2, wfrag);
    compute_w2_kernel<<<HEADS, 32, 0, stream>>>(w_rpe, w2);

    // 3) bitonic argsort of coords[:,0]
    sort_init_kernel<<<NTOK / 256, 256, 0, stream>>>(coords, keys, order);
    for (int k = 2; k <= NTOK; k <<= 1) {
        for (int j = k >> 1; j > 256; j >>= 1)
            bitonic_pass_kernel<<<NTOK / 256, 256, 0, stream>>>(keys, order, j, k);
        int jstart = (k >> 1) < 256 ? (k >> 1) : 256;
        bitonic_fused_kernel<<<NTOK / 512, 256, 0, stream>>>(keys, order, k, jstart);
    }
    gather_coords_kernel<<<NTOK / 256, 256, 0, stream>>>(order, coords, p_s);

    // 4) LN + QKV into sorted order
    ln_qkv_kernel<<<NTOK / 64, 256, 0, stream>>>(x, order, g1, be1, wfrag, q_s, k_s, vt);

    // 5) block-local attention
    attention_kernel<<<dim3(NB, HEADS), 128, 0, stream>>>(q_s, k_s, vt, p_s, w2, out_s);

    // 6) output projection + residual (scatter back to original order)
    out_proj_kernel<<<NTOK / 64, 128, 0, stream>>>(out_s, wfrag, b_out, x, order, x2);

    // 7) FFN + residual -> final output
    ffn_kernel<<<NTOK / 64, 128, 0, stream>>>(x2, g2, be2, wfrag, ff_b1, ff_b2, y);
}
